// YOLOv2OD_83708912599288
// MI455X (gfx1250) — compile-verified, matrix operand order checked by reading, not run on hardware
//
#include <hip/hip_runtime.h>

// ---------------------------------------------------------------------------
// YOLOv2 NMS post-processing for MI455X (gfx1250, wave32).
//   Kernel 1: per-candidate score/box/argmax + 64-bit stable sort key.
//   Kernel 2: per-image (1 workgroup): async-LDS key staging, bitonic sort,
//             greedy NMS, WMMA-based keep-flag block reduction, output write.
// Workspace use: 32 * 1445 * 32 B  ~= 1.5 MB.
// ---------------------------------------------------------------------------

#define NBOX   1445          // candidates per image (289 * 5)
#define NPAD   2048          // bitonic sort padding
#define NKPAD  1536          // keep-flag padding (3 * 512 for WMMA chunks)
#define MAXD   300
#define NCLS   80
#define HW     289
#define CCH    425
#define BS     512           // threads per block in kernel 2 (16 waves)
#define CONF_TH 0.6f
#define IOU_TH  0.45f

typedef __attribute__((ext_vector_type(16))) _Float16 v16h;
typedef __attribute__((ext_vector_type(8)))  float    v8f;
typedef __attribute__((ext_vector_type(2)))  int      v2i;

#if defined(__HIP_DEVICE_COMPILE__)
// ---- CDNA5 async global->LDS path (ASYNCcnt-tracked) ----------------------
// Builtin signature (from hipcc diagnostic): param0 = v2i addrspace(1)*,
// param1 = v2i addrspace(3)*, then constant offset + cpol.
#if __has_builtin(__builtin_amdgcn_global_load_async_to_lds_b64)
#define HAVE_ASYNC 1
#define ASYNC_B64(g, l)                                                       \
  __builtin_amdgcn_global_load_async_to_lds_b64(                              \
      (__attribute__((address_space(1))) v2i*)(unsigned long long)(g),        \
      (__attribute__((address_space(3))) v2i*)(unsigned)(unsigned long long)(l), \
      0, 0)
#else
#define HAVE_ASYNC 1
// flat LDS addresses carry the LDS byte offset in the low 32 bits
#define ASYNC_B64(g, l)                                                       \
  asm volatile("global_load_async_to_lds_b64 %0, %1, off"                     \
               :: "v"((unsigned)(unsigned long long)(l)),                     \
                  "v"((unsigned long long)(g)) : "memory")
#endif

__device__ __forceinline__ void wait_async0() {
#if __has_builtin(__builtin_amdgcn_s_wait_asynccnt)
  __builtin_amdgcn_s_wait_asynccnt(0);
#else
  asm volatile("s_wait_asynccnt 0" ::: "memory");
#endif
}
#endif // __HIP_DEVICE_COMPILE__

// ---------------------------------------------------------------------------
// Kernel 1: scores, boxes, argmax, sort keys.
// grid = (5 anchors, B images), block = 320 (10 waves), thread = spatial s.
// Reads are coalesced: consecutive threads read consecutive s of one channel.
// ---------------------------------------------------------------------------
__global__ __launch_bounds__(320) void yolo_scores_k1(
    const float* __restrict__ score,
    unsigned long long* __restrict__ wkey,
    float4* __restrict__ wbox,
    float* __restrict__ wconf,
    int* __restrict__ wcls)
{
  const int a = blockIdx.x;        // anchor 0..4
  const int b = blockIdx.y;        // image
  const int s = threadIdx.x;       // spatial 0..288
  if (s >= HW) return;

  const float* base = score + ((size_t)b * CCH + a * 85) * HW + s;
  const float cx  = base[0 * HW];
  const float cy  = base[1 * HW];
  const float w   = base[2 * HW];
  const float h   = base[3 * HW];
  const float obj = base[4 * HW];

  float conf = -3.402823466e+38f;
  int   j    = 0;
#pragma unroll 4
  for (int k = 0; k < NCLS; ++k) {
    const float c = base[(5 + k) * HW] * obj;   // cls = logits * obj
    if (c > conf) { conf = c; j = k; }          // first-max == jnp.argmax
  }

  const bool  valid = (obj > CONF_TH) && (conf > CONF_TH);
  const float sc    = valid ? conf : -1.0f;

  // order-preserving float->uint map, then invert for descending-score sort
  unsigned u = __float_as_uint(sc);
  u = (u & 0x80000000u) ? ~u : (u | 0x80000000u);   // ascending with sc
  const unsigned khi = ~u;                          // ascending == descending sc

  const int    n  = s * 5 + a;                      // candidate index
  const size_t gi = (size_t)b * NBOX + n;
  wkey[gi]  = ((unsigned long long)khi << 32) | (unsigned)n;  // stable tiebreak
  wbox[gi]  = make_float4(cx - w * 0.5f, cy - h * 0.5f,
                          cx + w * 0.5f, cy + h * 0.5f);
  wconf[gi] = conf;
  wcls[gi]  = j;
}

// ---------------------------------------------------------------------------
// Kernel 2: one workgroup per image. Sort + greedy NMS + compaction + output.
// ---------------------------------------------------------------------------
__global__ __launch_bounds__(BS) void yolo_nms_k2(
    const float* __restrict__ score,
    const unsigned long long* __restrict__ wkey,
    const float4* __restrict__ wbox,
    const float* __restrict__ wconf,
    const int* __restrict__ wcls,
    float* __restrict__ out)
{
  const int b   = blockIdx.x;
  const int tid = threadIdx.x;

  __shared__ unsigned long long sKey[NPAD];                 // 16 KB
  __shared__ float sX1[1456], sY1[1456], sX2[1456], sY2[1456], sA[1456];
  __shared__ int   sIdx[1456];
  __shared__ int   sKeep[NKPAD];
  __shared__ float sGsum[48];
  __shared__ int   sGpre[48];
  __shared__ int   sKeptR[MAXD];
  __shared__ int   sCount, sNumOut;

  const unsigned long long* kp = wkey + (size_t)b * NBOX;
  __builtin_prefetch(wbox + (size_t)b * NBOX, 0, 0);        // global_prefetch

  // ---- stage sort keys into LDS (async global->LDS path on gfx1250) ------
  for (int r = tid; r < NPAD; r += BS) {
    if (r < NBOX) {
#ifdef HAVE_ASYNC
      ASYNC_B64(kp + r, &sKey[r]);
#else
      sKey[r] = kp[r];
#endif
    } else {
      sKey[r] = 0xFFFFFFFFFFFFFFFFull;                      // pad sorts last
    }
  }
#ifdef HAVE_ASYNC
  wait_async0();
#endif
  __syncthreads();

  // ---- bitonic sort ascending over 2048 u64 keys --------------------------
  for (unsigned kk = 2; kk <= NPAD; kk <<= 1) {
    for (unsigned jj = kk >> 1; jj > 0; jj >>= 1) {
      for (int i = tid; i < NPAD; i += BS) {
        const unsigned l = (unsigned)i ^ jj;
        if (l > (unsigned)i) {
          const bool up = ((i & kk) == 0);
          const unsigned long long ka = sKey[i], kb = sKey[l];
          if ((up && ka > kb) || (!up && ka < kb)) {
            sKey[i] = kb; sKey[l] = ka;
          }
        }
      }
      __syncthreads();
    }
  }

  // ---- decode sorted order, build offset boxes, areas, keep flags ---------
  for (int r = tid; r < NBOX; r += BS) {
    const unsigned long long key = sKey[r];
    const int n = (int)(key & 0xFFFFFFFFu);
    sIdx[r] = n;
    // decode score to recover validity (invalid entries encode exactly -1.0f)
    const unsigned mapped = ~((unsigned)(key >> 32));
    const unsigned u0 = (mapped & 0x80000000u) ? (mapped ^ 0x80000000u) : ~mapped;
    const float sc = __uint_as_float(u0);

    const size_t gi = (size_t)b * NBOX + n;
    const float4 bx = wbox[gi];
    const float  off = (float)wcls[gi] * 4096.0f;           // j * MAX_WH
    const float x1 = bx.x + off, y1 = bx.y + off;
    const float x2 = bx.z + off, y2 = bx.w + off;
    sX1[r] = x1; sY1[r] = y1; sX2[r] = x2; sY2[r] = y2;
    sA[r]  = (x2 - x1) * (y2 - y1);                         // area on offset box
    sKeep[r] = (sc > -1.0f) ? 1 : 0;
  }
  for (int r = NBOX + tid; r < NKPAD; r += BS) sKeep[r] = 0;
  if (tid == 0) sCount = 0;
  __syncthreads();

  // ---- greedy sequential NMS (reference-equivalent) -----------------------
  for (int i = 0; i < NBOX; ++i) {
    if (*(volatile int*)&sCount >= MAXD) break;             // uniform after barrier
    const bool ki = (sKeep[i] != 0);
    if (ki) {
      const float ax1 = sX1[i], ay1 = sY1[i], ax2 = sX2[i], ay2 = sY2[i];
      const float aa  = sA[i];
      for (int r = i + 1 + tid; r < NBOX; r += BS) {
        if (!sKeep[r]) continue;
        const float xx1 = fmaxf(ax1, sX1[r]);
        const float yy1 = fmaxf(ay1, sY1[r]);
        const float xx2 = fminf(ax2, sX2[r]);
        const float yy2 = fminf(ay2, sY2[r]);
        const float ww = fmaxf(xx2 - xx1, 0.0f);
        const float hh = fmaxf(yy2 - yy1, 0.0f);
        const float inter = ww * hh;
        const float iou = inter / (aa + sA[r] - inter + 1e-9f);
        if (iou > IOU_TH) sKeep[r] = 0;
      }
    }
    if (ki && tid == 0) sCount++;
    __syncthreads();
  }
  __syncthreads();

  // ---- WMMA block reduction of keep flags: 512 flags -> 16 column sums ----
  // A = ones(16x32) f16, B[k][j] = keep flag -> D[:,j] = sum of 32 flags
  // (exact: 0/1 in f16, f32 accumulate). Columns j cover keep[base+32j..+31].
  {
    const int wave = tid >> 5, lane = tid & 31;
    if (wave < 3) {                                         // wave-uniform branch
      const int base = wave * 512;
      v16h av, bv;
#pragma unroll
      for (int t = 0; t < 16; ++t) av[t] = (_Float16)1.0f;
      const int rowbase = (lane < 16) ? (base + lane * 32)
                                      : (base + (lane - 16) * 32 + 16);
#pragma unroll
      for (int t = 0; t < 16; ++t)
        bv[t] = (_Float16)(float)sKeep[rowbase + t];
      v8f cv = {};
      v8f dv = __builtin_amdgcn_wmma_f32_16x16x32_f16(
          false, av, false, bv, (short)0, cv, false, false);
      if (lane < 16) sGsum[wave * 16 + lane] = dv[0];       // column sum
    }
  }
  __syncthreads();

  // ---- exclusive scan of 48 group sums ------------------------------------
  if (tid == 0) {
    int tot = 0;
    for (int g = 0; g < 48; ++g) {
      sGpre[g] = tot;
      tot += (int)(sGsum[g] + 0.5f);
    }
    sNumOut = (tot < MAXD) ? tot : MAXD;
  }
  __syncthreads();

  // ---- rank assignment & compaction ---------------------------------------
  for (int r = tid; r < NBOX; r += BS) {
    if (sKeep[r]) {
      const int g = r >> 5;
      int rank = sGpre[g];
      for (int m = (g << 5); m < r; ++m) rank += sKeep[m];
      if (rank < MAXD) sKeptR[rank] = r;
    }
  }
  __syncthreads();

  // ---- write output rows [box(4), conf, j, logits(80)]; zeros elsewhere ---
  float* ob = out + (size_t)b * MAXD * 86;
  const float* scb = score + (size_t)b * CCH * HW;
  for (int t = tid; t < MAXD * 86; t += BS) {
    const int rank = t / 86;
    const int k    = t - rank * 86;
    float v = 0.0f;
    if (rank < sNumOut) {
      const int r = sKeptR[rank];
      const int n = sIdx[r];
      const int a = n % 5;
      const int s = n / 5;
      const size_t gi = (size_t)b * NBOX + n;
      if (k < 4)       v = ((const float*)&wbox[gi])[k];
      else if (k == 4) v = wconf[gi];
      else if (k == 5) v = (float)wcls[gi];
      else             v = scb[(size_t)(a * 85 + 5 + (k - 6)) * HW + s];
    }
    ob[t] = v;
  }
}

// ---------------------------------------------------------------------------
extern "C" void kernel_launch(void* const* d_in, const int* in_sizes, int n_in,
                              void* d_out, int out_size, void* d_ws, size_t ws_size,
                              hipStream_t stream) {
  const float* score = (const float*)d_in[0];
  const int B = in_sizes[0] / (CCH * HW);     // 32

  char* ws = (char*)d_ws;
  const size_t nb = (size_t)B * NBOX;
  unsigned long long* wkey  = (unsigned long long*)(ws);
  float4*             wbox  = (float4*)(ws + nb * 8);
  float*              wconf = (float*)(ws + nb * 8 + nb * 16);
  int*                wcls  = (int*)(ws + nb * 8 + nb * 16 + nb * 4);

  yolo_scores_k1<<<dim3(5, B), dim3(320), 0, stream>>>(
      score, wkey, wbox, wconf, wcls);
  yolo_nms_k2<<<dim3(B), dim3(BS), 0, stream>>>(
      score, wkey, wbox, wconf, wcls, (float*)d_out);
}
